// ForcastConvTransformer_54546084659814
// MI455X (gfx1250) — matrix operand order repaired and checked by reading, not compiled, not run on hardware
//
#include <hip/hip_runtime.h>
#include <hip/hip_bf16.h>

typedef __attribute__((ext_vector_type(16))) _Float16 v16h;
typedef __attribute__((ext_vector_type(8)))  float    v8f;
typedef _Float16 half_t;

union V16H { v16h v; uint4 u[2]; _Float16 h[16]; };

__device__ __forceinline__ v8f wmma_acc(v16h a, v16h b, v8f c) {
  // D = A(16x32 f16) * B(32x16 f16) + C(16x16 f32)
  return __builtin_amdgcn_wmma_f32_16x16x32_f16(false, a, false, b, (short)0, c,
                                                false, false);
}

__device__ __forceinline__ float sigmoidf_(float x) { return 1.0f / (1.0f + __expf(-x)); }
__device__ __forceinline__ float siluf_(float x)    { return x * sigmoidf_(x); }

// ---------------------------------------------------------------------------
// Weight prep: fp32 -> f16 with layouts giving contiguous-K B operands.
// wqT/wkT: [j][c][f], wvT/wlT: [c][f], uniT: [n][K], w1T: [n][K], w2T: [n][K]
// ---------------------------------------------------------------------------
__global__ void k_prep(const float* wq, const float* wk, const float* wv,
                       const float* wl, const float* uni_w, const float* w1,
                       const float* w2,
                       half_t* wqT, half_t* wkT, half_t* wvT, half_t* wlT,
                       half_t* uniT, half_t* w1T, half_t* w2T) {
  int i = blockIdx.x * 256 + threadIdx.x;          // grid covers 5*512*64
  {
    int f = i & 63, c = (i >> 6) & 511, j = i >> 15;
    wqT[i] = (half_t)wq[(c * 64 + f) * 5 + j];
    wkT[i] = (half_t)wk[(c * 64 + f) * 5 + j];
  }
  if (i < 512 * 64) { wvT[i] = (half_t)wv[i]; wlT[i] = (half_t)wl[i]; }
  if (i < 64 * 1024) { int kk = i >> 10, K = i & 1023; uniT[i] = (half_t)uni_w[K * 64 + kk]; }
  if (i < 256 * 64)  { int n = i >> 6, K = i & 63;    w1T[i]  = (half_t)w1[K * 256 + n]; }
  if (i < 64 * 256)  { int n = i >> 8, K = i & 255;   w2T[i]  = (half_t)w2[K * 64 + n]; }
}

// ---------------------------------------------------------------------------
// Action stage 1: x3[b][c][t] = bn_g[c]*(x row . sq_w[c]) / sqrt(1+eps) + bn_b[c]
// ---------------------------------------------------------------------------
__global__ void k_x3(const float* x, const float* sqw, const float* bng,
                     const float* bnb, float* x3) {
  int i = blockIdx.x * 256 + threadIdx.x;          // b*32768 + c*1024 + tau
  int tau = i & 1023, c = (i >> 10) & 31, b = i >> 15;
  const float* xr = x + ((size_t)b * 1024 + tau) * 64;
  const float* w  = sqw + c * 64;
  float acc = 0.f;
  #pragma unroll 8
  for (int f = 0; f < 64; ++f) acc += w[f] * xr[f];
  x3[i] = bng[c] * acc * rsqrtf(1.0f + 1e-5f) + bnb[c];
}

// ---------------------------------------------------------------------------
// Action stage 2 + pos-emb: xin[b][t][k] (fp32 and f16). Block=64 per (b,tau).
// ---------------------------------------------------------------------------
__global__ void k_xin(const float* x, const float* x3, const float* dww,
                      const float* exw, const float* pos, float* xin,
                      half_t* xinh) {
  int bt = blockIdx.x;
  int tau = bt & 1023, b = bt >> 10;
  int kk = threadIdx.x;
  __shared__ float d[32];
  if (kk < 32) {
    int c = kk;
    const float* row = x3 + ((size_t)b * 32 + c) * 1024;
    float v0 = row[tau], dv;
    if (tau < 1023) {
      float v1 = row[tau + 1];
      float v2 = (tau + 2 < 1024) ? row[tau + 2] : 0.0f;
      dv = dww[c * 3 + 0] * v0 + dww[c * 3 + 1] * v1 + dww[c * 3 + 2] * v2 - v0;
    } else dv = v0;
    d[c] = dv;
  }
  __syncthreads();
  float p = 0.f;
  #pragma unroll 8
  for (int c = 0; c < 32; ++c) p += exw[kk * 32 + c] * d[c];
  size_t o = ((size_t)b * 1024 + tau) * 64 + kk;
  float v = x[o] * sigmoidf_(p) + pos[tau * 64 + kk];
  xin[o] = v; xinh[o] = (half_t)v;
}

// ---------------------------------------------------------------------------
// Projections via WMMA. One wave per (b, 16-row tile, 16-channel chunk).
// q/k: 5 dilated taps (row shift 2j-8), v/l: zero-shift tap.
// qg,kg: f16 [bh][t][64]; vgT: f16 [bh][k][t]; lo: f32 [bh][t][64]
// ---------------------------------------------------------------------------
__global__ void k_proj(const half_t* xinh, const half_t* wqT, const half_t* wkT,
                       const half_t* wvT, const half_t* wlT,
                       const float* bq, const float* bk,
                       half_t* qg, half_t* kg, half_t* vgT, float* lo) {
  int wid  = (blockIdx.x * blockDim.x + threadIdx.x) >> 5;
  int lane = threadIdx.x & 31;
  int nc = wid & 31, rt = (wid >> 5) & 63, b = wid >> 11;
  int r0 = rt * 16;
  int m = lane & 15, hf = lane >> 4;

  v8f accQ = {}, accK = {}, accV = {}, accL = {};
  for (int j = 0; j < 5; ++j) {
    int row = r0 + m + 2 * j - 8;
    bool valid = row >= 0;
    const half_t* arow = xinh + ((size_t)b * 1024 + row) * 64;
    #pragma unroll
    for (int fc = 0; fc < 2; ++fc) {
      V16H A;
      if (valid) {
        A.u[0] = *(const uint4*)(arow + fc * 32 + 16 * hf);
        A.u[1] = *(const uint4*)(arow + fc * 32 + 8 + 16 * hf);
      } else {
        A.u[0] = make_uint4(0u, 0u, 0u, 0u);
        A.u[1] = make_uint4(0u, 0u, 0u, 0u);
      }
      const half_t* bqp = wqT + ((size_t)j * 512 + nc * 16 + m) * 64 + fc * 32 + 16 * hf;
      V16H B;
      B.u[0] = *(const uint4*)bqp; B.u[1] = *(const uint4*)(bqp + 8);
      accQ = wmma_acc(A.v, B.v, accQ);
      const half_t* bkp = wkT + ((size_t)j * 512 + nc * 16 + m) * 64 + fc * 32 + 16 * hf;
      B.u[0] = *(const uint4*)bkp; B.u[1] = *(const uint4*)(bkp + 8);
      accK = wmma_acc(A.v, B.v, accK);
      if (j == 4) {  // shift-0 tap: 1x1 convs for v, l reuse A
        const half_t* bvp = wvT + ((size_t)(nc * 16 + m)) * 64 + fc * 32 + 16 * hf;
        B.u[0] = *(const uint4*)bvp; B.u[1] = *(const uint4*)(bvp + 8);
        accV = wmma_acc(A.v, B.v, accV);
        const half_t* blp = wlT + ((size_t)(nc * 16 + m)) * 64 + fc * 32 + 16 * hf;
        B.u[0] = *(const uint4*)blp; B.u[1] = *(const uint4*)(blp + 8);
        accL = wmma_acc(A.v, B.v, accL);
      }
    }
  }
  int n = lane & 15;
  int cg = nc * 16 + n;                     // conv channel
  int kk = cg >> 3, h = cg & 7;             // split_heads: c = kk*H + h
  int bh = b * 8 + h;
  float biasq = bq[cg], biask = bk[cg];
  const float sc = 0.35355339059327373f;    // 64^-0.25
  #pragma unroll
  for (int vv = 0; vv < 8; ++vv) {
    int tau = r0 + vv + 8 * hf;
    float q  = siluf_(accQ[vv] + biasq) * sc;
    float kv = siluf_(accK[vv] + biask) * sc;
    float vo = siluf_(accV[vv]);
    float lv = siluf_(accL[vv]);
    size_t idx = ((size_t)bh * 1024 + tau) * 64 + kk;
    qg[idx] = (half_t)q;
    kg[idx] = (half_t)kv;
    vgT[((size_t)bh * 64 + kk) * 1024 + tau] = (half_t)vo;
    lo[idx] = lv;
  }
}

// ---------------------------------------------------------------------------
// Local (banded, window=5, causal) attention: thread per (bh, tau), fp32 VALU.
// ---------------------------------------------------------------------------
__global__ void k_local(const float* lo, float* ol) {
  int i = blockIdx.x * 256 + threadIdx.x;   // bh*1024 + tau
  int tau = i & 1023, bh = i >> 10;
  const float* base = lo + (size_t)bh * 1024 * 64;
  const float* qrow = base + (size_t)tau * 64;
  float s[6]; float mx = -1e30f;
  int j0 = tau >= 5 ? 0 : 5 - tau;
  for (int j = j0; j < 6; ++j) {
    const float* kr = base + (size_t)(tau - 5 + j) * 64;
    float d = 0.f;
    #pragma unroll 8
    for (int f = 0; f < 64; ++f) d += qrow[f] * kr[f];
    d *= 0.125f; s[j] = d; mx = fmaxf(mx, d);
  }
  float sum = 0.f;
  for (int j = j0; j < 6; ++j) { s[j] = __expf(s[j] - mx); sum += s[j]; }
  float inv = 1.0f / sum;
  float out[64];
  #pragma unroll
  for (int f = 0; f < 64; ++f) out[f] = 0.f;
  for (int j = j0; j < 6; ++j) {
    const float* kr = base + (size_t)(tau - 5 + j) * 64;
    float p = s[j] * inv;
    #pragma unroll 8
    for (int f = 0; f < 64; ++f) out[f] += p * kr[f];
  }
  float* orow = ol + (size_t)bh * 1024 * 64 + (size_t)tau * 64;
  #pragma unroll 8
  for (int f = 0; f < 64; ++f) orow[f] = out[f];
}

// ---------------------------------------------------------------------------
// Causal flash attention, one wave per (bh, 16-row tile). og: f32 [bh][t][64].
// ---------------------------------------------------------------------------
__global__ void k_flash(const half_t* qg, const half_t* kg, const half_t* vgT,
                        float* og) {
  __shared__ _Float16 Pl[8][512];           // per-wave 16x32 P tile
  int wblk = threadIdx.x >> 5;
  int wid  = (blockIdx.x * blockDim.x + threadIdx.x) >> 5;
  int lane = threadIdx.x & 31;
  int rt = wid & 63, bh = wid >> 6;
  int r0 = rt * 16;
  int m = lane & 15, hf = lane >> 4, n = lane & 15;
  _Float16* P = Pl[wblk];

  const half_t* qrow = qg + ((size_t)bh * 1024 + r0 + m) * 64;
  V16H A0, A1;
  A0.u[0] = *(const uint4*)(qrow + 16 * hf);
  A0.u[1] = *(const uint4*)(qrow + 8 + 16 * hf);
  A1.u[0] = *(const uint4*)(qrow + 32 + 16 * hf);
  A1.u[1] = *(const uint4*)(qrow + 40 + 16 * hf);

  v8f zero = {};
  v8f Oa[4] = {zero, zero, zero, zero};
  float mrun[8], lrun[8];
  #pragma unroll
  for (int vv = 0; vv < 8; ++vv) { mrun[vv] = -1e30f; lrun[vv] = 0.f; }

  int ntiles = (r0 + 16 + 31) >> 5;
  for (int ct = 0; ct < ntiles; ++ct) {
    int c0 = ct * 32;
    v8f S0 = {}, S1 = {};
    const half_t* kc0 = kg + ((size_t)bh * 1024 + c0 + n) * 64;
    const half_t* kc1 = kc0 + 16 * 64;
    V16H B;
    B.u[0] = *(const uint4*)(kc0 + 16 * hf);      B.u[1] = *(const uint4*)(kc0 + 16 * hf + 8);
    S0 = wmma_acc(A0.v, B.v, S0);
    B.u[0] = *(const uint4*)(kc0 + 32 + 16 * hf); B.u[1] = *(const uint4*)(kc0 + 40 + 16 * hf);
    S0 = wmma_acc(A1.v, B.v, S0);
    B.u[0] = *(const uint4*)(kc1 + 16 * hf);      B.u[1] = *(const uint4*)(kc1 + 16 * hf + 8);
    S1 = wmma_acc(A0.v, B.v, S1);
    B.u[0] = *(const uint4*)(kc1 + 32 + 16 * hf); B.u[1] = *(const uint4*)(kc1 + 40 + 16 * hf);
    S1 = wmma_acc(A1.v, B.v, S1);

    // causal mask + per-row max over the 32 columns
    int col0 = c0 + n, col1 = col0 + 16;
    float tmax[8];
    #pragma unroll
    for (int vv = 0; vv < 8; ++vv) {
      int row = r0 + vv + 8 * hf;
      float s0 = (col0 <= row) ? S0[vv] : -1e30f;
      float s1 = (col1 <= row) ? S1[vv] : -1e30f;
      S0[vv] = s0; S1[vv] = s1;
      tmax[vv] = fmaxf(s0, s1);
    }
    #pragma unroll
    for (int off = 1; off < 16; off <<= 1)
      #pragma unroll
      for (int vv = 0; vv < 8; ++vv)
        tmax[vv] = fmaxf(tmax[vv], __shfl_xor(tmax[vv], off));
    float alpha[8], psum[8];
    #pragma unroll
    for (int vv = 0; vv < 8; ++vv) {
      float mn = fmaxf(mrun[vv], tmax[vv]);
      alpha[vv] = __expf(mrun[vv] - mn);
      mrun[vv] = mn;
      float p0 = __expf(S0[vv] - mn);
      float p1 = __expf(S1[vv] - mn);
      S0[vv] = p0; S1[vv] = p1;
      psum[vv] = p0 + p1;
    }
    #pragma unroll
    for (int off = 1; off < 16; off <<= 1)
      #pragma unroll
      for (int vv = 0; vv < 8; ++vv)
        psum[vv] += __shfl_xor(psum[vv], off);
    #pragma unroll
    for (int vv = 0; vv < 8; ++vv) {
      lrun[vv] = lrun[vv] * alpha[vv] + psum[vv];
      Oa[0][vv] *= alpha[vv]; Oa[1][vv] *= alpha[vv];
      Oa[2][vv] *= alpha[vv]; Oa[3][vv] *= alpha[vv];
    }

    // P: C-layout -> A-layout via LDS (wave-local)
    #pragma unroll
    for (int vv = 0; vv < 8; ++vv) {
      int row = vv + 8 * hf;
      P[row * 32 + n]      = (_Float16)S0[vv];
      P[row * 32 + 16 + n] = (_Float16)S1[vv];
    }
    asm volatile("s_wait_dscnt 0" ::: "memory");
    __builtin_amdgcn_wave_barrier();
    V16H PA;
    PA.u[0] = *(const uint4*)(P + m * 32 + 16 * hf);
    PA.u[1] = *(const uint4*)(P + m * 32 + 8 + 16 * hf);

    #pragma unroll
    for (int kc = 0; kc < 4; ++kc) {
      const half_t* vp = vgT + ((size_t)bh * 64 + kc * 16 + n) * 1024 + c0 + 16 * hf;
      V16H Bv; Bv.u[0] = *(const uint4*)vp; Bv.u[1] = *(const uint4*)(vp + 8);
      Oa[kc] = wmma_acc(PA.v, Bv.v, Oa[kc]);
    }
    asm volatile("s_wait_dscnt 0" ::: "memory");
    __builtin_amdgcn_wave_barrier();
  }

  #pragma unroll
  for (int vv = 0; vv < 8; ++vv) {
    float inv = 1.0f / lrun[vv];
    int tau = r0 + vv + 8 * hf;
    float* orow = og + ((size_t)bh * 1024 + tau) * 64;
    orow[n]      = Oa[0][vv] * inv;
    orow[16 + n] = Oa[1][vv] * inv;
    orow[32 + n] = Oa[2][vv] * inv;
    orow[48 + n] = Oa[3][vv] * inv;
  }
}

// ---------------------------------------------------------------------------
// Gate + concat -> out_f16 [b][t][1024]
// ---------------------------------------------------------------------------
__global__ void k_combine(const float* og, const float* ol, half_t* outh) {
  int i = blockIdx.x * 256 + threadIdx.x;   // bh*65536 + tau*64 + kk
  int kk = i & 63, tau = (i >> 6) & 1023, bh = i >> 16;
  int h = bh & 7, b = bh >> 3;
  float g = og[i], l = ol[i];
  float gate = sigmoidf_(g);
  size_t base = ((size_t)b * 1024 + tau) * 1024;
  outh[base + h * 64 + kk]       = (half_t)((1.f - gate) * l);
  outh[base + 512 + h * 64 + kk] = (half_t)(gate * g);
}

// ---------------------------------------------------------------------------
// uni projection (K=1024) + bias + silu + residual -> yres f32
// ---------------------------------------------------------------------------
__global__ void k_uni(const half_t* outh, const half_t* uniT, const float* unib,
                      const float* xin, float* yres) {
  int wid  = (blockIdx.x * blockDim.x + threadIdx.x) >> 5;
  int lane = threadIdx.x & 31;
  int nc = wid & 3, rt = (wid >> 2) & 63, b = wid >> 8;
  int r0 = rt * 16, m = lane & 15, hf = lane >> 4;
  const half_t* arow = outh + ((size_t)b * 1024 + r0 + m) * 1024;
  v8f acc = {};
  for (int kc = 0; kc < 32; ++kc) {
    V16H A;
    A.u[0] = *(const uint4*)(arow + kc * 32 + 16 * hf);
    A.u[1] = *(const uint4*)(arow + kc * 32 + 8 + 16 * hf);
    const half_t* bp = uniT + ((size_t)(nc * 16 + m)) * 1024 + kc * 32 + 16 * hf;
    V16H B; B.u[0] = *(const uint4*)bp; B.u[1] = *(const uint4*)(bp + 8);
    acc = wmma_acc(A.v, B.v, acc);
  }
  int n = lane & 15, kk = nc * 16 + n;
  float bias = unib[kk];
  #pragma unroll
  for (int vv = 0; vv < 8; ++vv) {
    int tau = r0 + vv + 8 * hf;
    float u = siluf_(acc[vv] + bias);
    size_t idx = ((size_t)b * 1024 + tau) * 64 + kk;
    yres[idx] = u + xin[idx];
  }
}

// ---------------------------------------------------------------------------
// LayerNorm over last dim (64). Block=64 per (b,t). Optional f16 mirror.
// ---------------------------------------------------------------------------
__global__ void k_ln(const float* in, const float* g, const float* bb,
                     float* out, half_t* outh) {
  int bt = blockIdx.x, kk = threadIdx.x;
  __shared__ float sm[64];
  float v = in[(size_t)bt * 64 + kk];
  sm[kk] = v; __syncthreads();
  for (int s = 32; s > 0; s >>= 1) { if (kk < s) sm[kk] += sm[kk + s]; __syncthreads(); }
  float mean = sm[0] * (1.0f / 64.0f);
  __syncthreads();
  float d = v - mean;
  sm[kk] = d * d; __syncthreads();
  for (int s = 32; s > 0; s >>= 1) { if (kk < s) sm[kk] += sm[kk + s]; __syncthreads(); }
  float var = sm[0] * (1.0f / 64.0f);
  float r = d * rsqrtf(var + 1e-5f) * g[kk] + bb[kk];
  out[(size_t)bt * 64 + kk] = r;
  if (outh) outh[(size_t)bt * 64 + kk] = (half_t)r;
}

// ---------------------------------------------------------------------------
// FF1: [4096x64]@[64x256] + bias + relu -> h f16
// ---------------------------------------------------------------------------
__global__ void k_ff1(const half_t* yh, const half_t* w1T, const float* b1,
                      half_t* hh) {
  int wid  = (blockIdx.x * blockDim.x + threadIdx.x) >> 5;
  int lane = threadIdx.x & 31;
  int nc = wid & 15, rt = (wid >> 4) & 63, b = wid >> 10;
  int r0 = rt * 16, m = lane & 15, hf = lane >> 4;
  const half_t* arow = yh + ((size_t)b * 1024 + r0 + m) * 64;
  v8f acc = {};
  #pragma unroll
  for (int kc = 0; kc < 2; ++kc) {
    V16H A;
    A.u[0] = *(const uint4*)(arow + kc * 32 + 16 * hf);
    A.u[1] = *(const uint4*)(arow + kc * 32 + 8 + 16 * hf);
    const half_t* bp = w1T + ((size_t)(nc * 16 + m)) * 64 + kc * 32 + 16 * hf;
    V16H B; B.u[0] = *(const uint4*)bp; B.u[1] = *(const uint4*)(bp + 8);
    acc = wmma_acc(A.v, B.v, acc);
  }
  int n = lane & 15, col = nc * 16 + n;
  float bias = b1[col];
  #pragma unroll
  for (int vv = 0; vv < 8; ++vv) {
    int tau = r0 + vv + 8 * hf;
    float u = fmaxf(acc[vv] + bias, 0.0f);
    hh[((size_t)b * 1024 + tau) * 256 + col] = (half_t)u;
  }
}

// ---------------------------------------------------------------------------
// FF2: [4096x256]@[256x64] + bias + residual(y) -> z f32
// ---------------------------------------------------------------------------
__global__ void k_ff2(const half_t* hh, const half_t* w2T, const float* b2,
                      const float* y, float* z) {
  int wid  = (blockIdx.x * blockDim.x + threadIdx.x) >> 5;
  int lane = threadIdx.x & 31;
  int nc = wid & 3, rt = (wid >> 2) & 63, b = wid >> 8;
  int r0 = rt * 16, m = lane & 15, hf = lane >> 4;
  const half_t* arow = hh + ((size_t)b * 1024 + r0 + m) * 256;
  v8f acc = {};
  #pragma unroll
  for (int kc = 0; kc < 8; ++kc) {
    V16H A;
    A.u[0] = *(const uint4*)(arow + kc * 32 + 16 * hf);
    A.u[1] = *(const uint4*)(arow + kc * 32 + 8 + 16 * hf);
    const half_t* bp = w2T + ((size_t)(nc * 16 + m)) * 256 + kc * 32 + 16 * hf;
    V16H B; B.u[0] = *(const uint4*)bp; B.u[1] = *(const uint4*)(bp + 8);
    acc = wmma_acc(A.v, B.v, acc);
  }
  int n = lane & 15, kk = nc * 16 + n;
  float bias = b2[kk];
  #pragma unroll
  for (int vv = 0; vv < 8; ++vv) {
    int tau = r0 + vv + 8 * hf;
    size_t idx = ((size_t)b * 1024 + tau) * 64 + kk;
    z[idx] = acc[vv] + bias + y[idx];
  }
}

// ---------------------------------------------------------------------------
extern "C" void kernel_launch(void* const* d_in, const int* in_sizes, int n_in,
                              void* d_out, int out_size, void* d_ws, size_t ws_size,
                              hipStream_t stream) {
  (void)in_sizes; (void)n_in; (void)out_size; (void)ws_size;
  const float* x     = (const float*)d_in[0];
  const float* sq_w  = (const float*)d_in[1];
  const float* bn_g  = (const float*)d_in[2];
  const float* bn_b  = (const float*)d_in[3];
  const float* dw_w  = (const float*)d_in[4];
  const float* ex_w  = (const float*)d_in[5];
  const float* pos   = (const float*)d_in[6];
  const float* wq    = (const float*)d_in[7];
  const float* bq    = (const float*)d_in[8];
  const float* wk    = (const float*)d_in[9];
  const float* bk    = (const float*)d_in[10];
  const float* wv    = (const float*)d_in[11];
  const float* wl    = (const float*)d_in[12];
  const float* uni_w = (const float*)d_in[13];
  const float* uni_b = (const float*)d_in[14];
  const float* ln1_g = (const float*)d_in[15];
  const float* ln1_b = (const float*)d_in[16];
  const float* ln2_g = (const float*)d_in[17];
  const float* ln2_b = (const float*)d_in[18];
  const float* ff_w1 = (const float*)d_in[19];
  const float* ff_b1 = (const float*)d_in[20];
  const float* ff_w2 = (const float*)d_in[21];
  const float* ff_b2 = (const float*)d_in[22];
  float* out = (float*)d_out;

  char* wp = (char*)d_ws;
  auto alloc = [&](size_t bytes) -> void* {
    void* p = (void*)wp; wp += (bytes + 255) & ~(size_t)255; return p;
  };
  float*  xin  = (float*) alloc((size_t)4 * 1024 * 64 * 4);
  half_t* xinh = (half_t*)alloc((size_t)4 * 1024 * 64 * 2);
  float*  x3   = (float*) alloc((size_t)4 * 32 * 1024 * 4);
  half_t* qg   = (half_t*)alloc((size_t)32 * 1024 * 64 * 2);
  half_t* kgb  = (half_t*)alloc((size_t)32 * 1024 * 64 * 2);
  half_t* vgT  = (half_t*)alloc((size_t)32 * 64 * 1024 * 2);
  float*  lo   = (float*) alloc((size_t)32 * 1024 * 64 * 4);
  float*  og   = (float*) alloc((size_t)32 * 1024 * 64 * 4);
  float*  ol   = (float*) alloc((size_t)32 * 1024 * 64 * 4);
  half_t* outh = (half_t*)alloc((size_t)4 * 1024 * 1024 * 2);
  float*  yres = (float*) alloc((size_t)4 * 1024 * 64 * 4);
  float*  yln  = (float*) alloc((size_t)4 * 1024 * 64 * 4);
  half_t* ylnh = (half_t*)alloc((size_t)4 * 1024 * 64 * 2);
  half_t* hh   = (half_t*)alloc((size_t)4 * 1024 * 256 * 2);
  float*  z    = (float*) alloc((size_t)4 * 1024 * 64 * 4);
  half_t* wqT  = (half_t*)alloc((size_t)5 * 512 * 64 * 2);
  half_t* wkT  = (half_t*)alloc((size_t)5 * 512 * 64 * 2);
  half_t* wvT  = (half_t*)alloc((size_t)512 * 64 * 2);
  half_t* wlT  = (half_t*)alloc((size_t)512 * 64 * 2);
  half_t* uniT = (half_t*)alloc((size_t)64 * 1024 * 2);
  half_t* w1T  = (half_t*)alloc((size_t)256 * 64 * 2);
  half_t* w2T  = (half_t*)alloc((size_t)64 * 256 * 2);

  k_prep   <<<640,  256, 0, stream>>>(wq, wk, wv, wl, uni_w, ff_w1, ff_w2,
                                      wqT, wkT, wvT, wlT, uniT, w1T, w2T);
  k_x3     <<<512,  256, 0, stream>>>(x, sq_w, bn_g, bn_b, x3);
  k_xin    <<<4096, 64,  0, stream>>>(x, x3, dw_w, ex_w, pos, xin, xinh);
  k_proj   <<<1024, 256, 0, stream>>>(xinh, wqT, wkT, wvT, wlT, bq, bk,
                                      qg, kgb, vgT, lo);
  k_local  <<<128,  256, 0, stream>>>(lo, ol);
  k_flash  <<<256,  256, 0, stream>>>(qg, kgb, vgT, og);
  k_combine<<<8192, 256, 0, stream>>>(og, ol, outh);
  k_uni    <<<128,  256, 0, stream>>>(outh, uniT, uni_b, xin, yres);
  k_ln     <<<4096, 64,  0, stream>>>(yres, ln1_g, ln1_b, yln, ylnh);
  k_ff1    <<<512,  256, 0, stream>>>(ylnh, w1T, ff_b1, hh);
  k_ff2    <<<128,  256, 0, stream>>>(hh, w2T, ff_b2, yln, z);
  k_ln     <<<4096, 64,  0, stream>>>(z, ln2_g, ln2_b, out, (half_t*)nullptr);
}
